// TopKRouter_36240934043696
// MI455X (gfx1250) — compile-verified, matrix operand order checked by reading, not run on hardware
//
#include <hip/hip_runtime.h>
#include <hip/hip_bf16.h>
#include <math.h>

// ---------------------------------------------------------------------------
// TopK MoE router for MI455X (gfx1250, wave32, WMMA).
//   inputs: [4,4096,4096] f32   W: [4096,64] f32
//   out: weights[16384*2] f32 | indices[16384*2] (as f32) | loss (1 f32)
// Gate GEMM via V_WMMA_F32_16X16X32_BF16 (bf16 in, f32 acc) — memory-bound
// at ~268 MB / 23.3 TB/s ≈ 11.5 us.
// ---------------------------------------------------------------------------

typedef __attribute__((ext_vector_type(16))) __bf16 v16bf;
typedef __attribute__((ext_vector_type(8)))  float  v8f;

#define DDIM    4096
#define NEXP    64
#define NTOK    16384          // 4*4096
#define KCHUNKS 128            // DDIM / 32
#define NTILES  1024           // NTOK / 16
#define WAVES_PER_BLK 4

// ws layout: [0,516)   float accum[129]  (0: sum_lse2, 1..64: counts, 65..128: sumprobs)
//            [1024, 1024+512K) bf16 Wb, packed WMMA-B layout
#define WB_OFS 1024

// --------------------------- accumulator zeroing ---------------------------
__global__ void zero_accum(float* accum) {
    int t = threadIdx.x;
    if (t < 129) accum[t] = 0.0f;
}

// ----------------------- pack W into WMMA B layout -------------------------
// Wb index = (((ntile*128 + chunk)*32 + lane)*16 + j)
// lane<16 : N = ntile*16 + lane,      K = chunk*32 + (j<8 ? j : j+8)
// lane>=16: N = ntile*16 + (lane-16), K = chunk*32 + 8 + (j<8 ? j : j+8)
__global__ void pack_w(const float* __restrict__ W, __hip_bfloat16* __restrict__ Wb) {
    int t = blockIdx.x * blockDim.x + threadIdx.x;   // 0 .. 4*128*32-1
    if (t >= 4 * KCHUNKS * 32) return;
    int lane = t & 31;
    int ch   = (t >> 5) & (KCHUNKS - 1);
    int nt   = t >> 12;
    int n    = nt * 16 + (lane & 15);
    int kofs = (lane >> 4) * 8;
    __hip_bfloat16* dst = Wb + (size_t)t * 16;
    #pragma unroll
    for (int j = 0; j < 16; ++j) {
        int k = ch * 32 + kofs + (j < 8 ? j : j + 8);
        dst[j] = __float2bfloat16(W[(size_t)k * NEXP + n]);
    }
}

// ------------------------------- main kernel -------------------------------
__global__ __launch_bounds__(32 * WAVES_PER_BLK)
void router_main(const float* __restrict__ inp,
                 const __hip_bfloat16* __restrict__ Wb,
                 float* __restrict__ out,
                 float* __restrict__ accum) {
    __shared__ float tileL[WAVES_PER_BLK][16 * NEXP];   // 16 KB
    __shared__ float red[WAVES_PER_BLK][16];

    const int wave = threadIdx.x >> 5;
    const int lane = threadIdx.x & 31;
    const int tile = blockIdx.x * WAVES_PER_BLK + wave;   // 0..1023
    const int row  = lane & 15;
    const int hi   = lane >> 4;                            // 0 or 1

    // A operand: this lane streams token row (tile*16+row),
    // K runs {c+kofs .. c+kofs+7} and {c+kofs+16 .. c+kofs+23}, kofs = hi*8.
    const float* arow = inp + (size_t)(tile * 16 + row) * DDIM + hi * 8;
    const __hip_bfloat16* bbase = Wb + (size_t)lane * 16;

    v8f c0 = {}, c1 = {}, c2 = {}, c3 = {};

    for (int ch = 0; ch < KCHUNKS; ++ch) {
        const float4* p = (const float4*)(arow + ch * 32);
        float4 x0 = p[0];
        float4 x1 = p[1];
        float4 x2 = p[4];   // +16 floats
        float4 x3 = p[5];

        v16bf a;
        a[0]  = (__bf16)x0.x;  a[1]  = (__bf16)x0.y;
        a[2]  = (__bf16)x0.z;  a[3]  = (__bf16)x0.w;
        a[4]  = (__bf16)x1.x;  a[5]  = (__bf16)x1.y;
        a[6]  = (__bf16)x1.z;  a[7]  = (__bf16)x1.w;
        a[8]  = (__bf16)x2.x;  a[9]  = (__bf16)x2.y;
        a[10] = (__bf16)x2.z;  a[11] = (__bf16)x2.w;
        a[12] = (__bf16)x3.x;  a[13] = (__bf16)x3.y;
        a[14] = (__bf16)x3.z;  a[15] = (__bf16)x3.w;

        // B operands: contiguous 32 B per lane, L2-resident.
        const size_t chBase = (size_t)ch * 32 * 16;
        const size_t ntStep = (size_t)KCHUNKS * 32 * 16;
        v16bf b0 = *reinterpret_cast<const v16bf*>(bbase + chBase + 0 * ntStep);
        v16bf b1 = *reinterpret_cast<const v16bf*>(bbase + chBase + 1 * ntStep);
        v16bf b2 = *reinterpret_cast<const v16bf*>(bbase + chBase + 2 * ntStep);
        v16bf b3 = *reinterpret_cast<const v16bf*>(bbase + chBase + 3 * ntStep);

        c0 = __builtin_amdgcn_wmma_f32_16x16x32_bf16(false, a, false, b0, (short)0, c0, false, false);
        c1 = __builtin_amdgcn_wmma_f32_16x16x32_bf16(false, a, false, b1, (short)0, c1, false, false);
        c2 = __builtin_amdgcn_wmma_f32_16x16x32_bf16(false, a, false, b2, (short)0, c2, false, false);
        c3 = __builtin_amdgcn_wmma_f32_16x16x32_bf16(false, a, false, b3, (short)0, c3, false, false);
    }

    // Spill logits tile to LDS. C/D layout: vgpr r, lane<16 -> (M=r, N=lane),
    // lane>=16 -> (M=r+8, N=lane-16).
    float* T = tileL[wave];
    #pragma unroll
    for (int r = 0; r < 8; ++r) {
        int m = r + hi * 8;
        T[m * NEXP +  0 + row] = c0[r];
        T[m * NEXP + 16 + row] = c1[r];
        T[m * NEXP + 32 + row] = c2[r];
        T[m * NEXP + 48 + row] = c3[r];
    }
    __syncthreads();

    // Per-token softmax / top-2 / outputs (lanes 0..15, one token each).
    if (lane < 16) {
        float* L = T + row * NEXP;
        float mx = L[0];
        #pragma unroll 8
        for (int e = 1; e < NEXP; ++e) mx = fmaxf(mx, L[e]);
        float s = 0.0f;
        #pragma unroll 8
        for (int e = 0; e < NEXP; ++e) s += __expf(L[e] - mx);
        float inv = 1.0f / s;

        float v0 = -INFINITY, v1 = -INFINITY;
        int   i0 = 0,          i1 = 0;
        for (int e = 0; e < NEXP; ++e) {
            float x = L[e];
            if (x > v0)      { v1 = v0; i1 = i0; v0 = x; i0 = e; }
            else if (x > v1) { v1 = x; i1 = e; }
        }
        float p0 = __expf(v0 - mx) * inv;
        float p1 = __expf(v1 - mx) * inv;
        float dn = 1.0f / (p0 + p1 + 1e-9f);

        int tok = tile * 16 + row;
        out[(size_t)tok * 2 + 0] = p0 * dn;
        out[(size_t)tok * 2 + 1] = p1 * dn;
        out[2 * NTOK + (size_t)tok * 2 + 0] = (float)i0;
        out[2 * NTOK + (size_t)tok * 2 + 1] = (float)i1;

        atomicAdd(accum + 1 + i0, 1.0f);     // expert counts (top-2 distinct)
        atomicAdd(accum + 1 + i1, 1.0f);

        float lse = mx + __logf(s);
        red[wave][row] = lse * lse;

        // overwrite logits with probs for the column reduction
        #pragma unroll 8
        for (int e = 0; e < NEXP; ++e) L[e] = __expf(L[e] - mx) * inv;
    }
    __syncthreads();

    if (lane == 0) {
        float s = 0.0f;
        #pragma unroll
        for (int i = 0; i < 16; ++i) s += red[wave][i];
        atomicAdd(accum + 0, s);
    }
    // sum of probs per expert over this wave's 16 tokens; 2 experts per lane
    float s1 = 0.0f, s2 = 0.0f;
    #pragma unroll
    for (int r = 0; r < 16; ++r) {
        s1 += T[r * NEXP + lane];
        s2 += T[r * NEXP + lane + 32];
    }
    atomicAdd(accum + 65 + lane,      s1);
    atomicAdd(accum + 65 + lane + 32, s2);
}

// ------------------------------- loss finalize -----------------------------
__global__ void finalize(const float* __restrict__ accum, float* __restrict__ out) {
    __shared__ float sh[NEXP];
    int t = threadIdx.x;
    sh[t] = accum[1 + t] * accum[65 + t];   // counts[e] * sumprobs[e]
    __syncthreads();
    if (t == 0) {
        float dot = 0.0f;
        for (int i = 0; i < NEXP; ++i) dot += sh[i];
        const float N = (float)NTOK;
        out[4 * NTOK] = accum[0] / N + (float)NEXP * dot / (N * N);
    }
}

// ------------------------------- launcher ----------------------------------
extern "C" void kernel_launch(void* const* d_in, const int* in_sizes, int n_in,
                              void* d_out, int out_size, void* d_ws, size_t ws_size,
                              hipStream_t stream) {
    const float* inp = (const float*)d_in[0];   // [4,4096,4096] f32
    const float* W   = (const float*)d_in[1];   // [4096,64] f32
    float* out   = (float*)d_out;
    float* accum = (float*)d_ws;
    __hip_bfloat16* Wb = (__hip_bfloat16*)((char*)d_ws + WB_OFS);

    zero_accum<<<1, 256, 0, stream>>>(accum);
    pack_w<<<(4 * KCHUNKS * 32 + 255) / 256, 256, 0, stream>>>(W, Wb);
    router_main<<<NTILES / WAVES_PER_BLK, 32 * WAVES_PER_BLK, 0, stream>>>(inp, Wb, out, accum);
    finalize<<<1, NEXP, 0, stream>>>(accum, out);
}
